// SpGraphAttentionLayer_e2t_37641093382714
// MI455X (gfx1250) — matrix operand order repaired and checked by reading, not compiled, not run on hardware
//
#include <hip/hip_runtime.h>

#define N1 100000
#define N2 500
#define NE 1000000
#define EIN 128
#define TIN 128
#define NREL 64
#define KTOT 320
#define OUTF 128
#define ALPHA 0.2f

typedef __attribute__((ext_vector_type(16))) __bf16         v16bf;
typedef __attribute__((ext_vector_type(2)))  __bf16         v2bf;
typedef __attribute__((ext_vector_type(2)))  float          f2;
typedef __attribute__((ext_vector_type(8)))  float          v8f;
typedef __attribute__((ext_vector_type(8)))  unsigned short us8;
typedef __attribute__((ext_vector_type(4)))  float          f4;

union FragU { v16bf bf; us8 u[2]; v2bf p[8]; };

__device__ __forceinline__ unsigned short f2bf(float f) {
  union { float f; unsigned u; } c; c.f = f;
  unsigned u = c.u;
  unsigned r = (u + 0x7FFFu + ((u >> 16) & 1u)) >> 16;  // RNE
  return (unsigned short)r;
}

// Vector fptrunc -> v_cvt_pk_bf16_f32 on gfx1250 (confirmed by disasm).
__device__ __forceinline__ v2bf pack_bf16(float lo, float hi) {
  f2 x = {lo, hi};
  return __builtin_convertvector(x, v2bf);
}

// ---------------------------------------------------------------------------
// Prep: swizzle a[128][320] (f32) into bf16 B-fragments, step-major order:
// frag step = kk*8 + t, lane, element e:
//   value = a[t*16 + (lane&15)][kk*32 + (lane&16) + e]
// ---------------------------------------------------------------------------
__global__ void prep_bfrag(const float* __restrict__ a,
                           unsigned short* __restrict__ bws) {
  int idx = blockIdx.x * 256 + threadIdx.x;
  if (idx >= 80 * 32 * 16) return;
  int e    = idx & 15;
  int lane = (idx >> 4) & 31;
  int f    = idx >> 9;            // step = kk*8 + t
  int t    = f & 7;
  int kk   = f >> 3;
  int n = t * 16 + (lane & 15);
  int k = kk * 32 + (lane & 16) + e;
  bws[idx] = f2bf(a[n * KTOT + k]);
}

// Build one 16x32 bf16 A fragment for this lane directly from global memory.
// Lane supplies row M = lane&15; two 8-aligned K-runs never straddle the
// x1 / x2 / edge_embed region boundaries (128, 256).
__device__ __forceinline__ FragU buildA(const float* __restrict__ x1,
                                        const float* __restrict__ x2,
                                        const float* __restrict__ eemb,
                                        int src, int dst, int edge,
                                        int kk, int half) {
  FragU A;
  #pragma unroll
  for (int run = 0; run < 2; ++run) {
    int k0 = kk * 32 + half * 16 + run * 8;
    const float* p;
    if (k0 < 128)      p = x1   + (size_t)src  * EIN + k0;
    else if (k0 < 256) p = x2   + (size_t)dst  * TIN + (k0 - 128);
    else               p = eemb + (size_t)edge * NREL + (k0 - 256);
    f4 q0 = *(const f4*)p;
    f4 q1 = *(const f4*)(p + 4);
    A.p[run * 4 + 0] = pack_bf16(q0.x, q0.y);
    A.p[run * 4 + 1] = pack_bf16(q0.z, q0.w);
    A.p[run * 4 + 2] = pack_bf16(q1.x, q1.y);
    A.p[run * 4 + 3] = pack_bf16(q1.z, q1.w);
  }
  return A;
}

__device__ __forceinline__ void loadB(FragU& B, const unsigned short* sBlane,
                                      int step) {
  const us8* bp = (const us8*)(sBlane + (size_t)step * 512);
  B.u[0] = bp[0];
  B.u[1] = bp[1];
}

// Attention scalar + weighted scatter for one 16-edge sub-tile.
__device__ __forceinline__ void epilogue(const v8f* __restrict__ acc,
                                         const float* __restrict__ a2v,
                                         int base, int half, int lx,
                                         const int* __restrict__ esrc,
                                         const int* __restrict__ edst,
                                         float* __restrict__ ent_num,
                                         float* __restrict__ ent_den,
                                         float* __restrict__ typ_num,
                                         float* __restrict__ typ_den) {
  float ee[8];
  #pragma unroll
  for (int i = 0; i < 8; ++i) {
    float s = 0.f;
    #pragma unroll
    for (int t = 0; t < 8; ++t) s += acc[t][i] * a2v[t];
    s += __shfl_xor(s, 1, 32);
    s += __shfl_xor(s, 2, 32);
    s += __shfl_xor(s, 4, 32);
    s += __shfl_xor(s, 8, 32);
    float lr = s > 0.f ? s : ALPHA * s;
    ee[i] = __expf(-lr);
  }

  int srcm[8], dstm[8];
  #pragma unroll
  for (int i = 0; i < 8; ++i) {
    int m   = half * 8 + i;            // C/D layout: row = i + 8*half
    srcm[i] = esrc[base + m];
    dstm[i] = edst[base + m];
  }

  #pragma unroll
  for (int t = 0; t < 8; ++t) {
    int nn = t * 16 + lx;
    #pragma unroll
    for (int i = 0; i < 8; ++i) {
      float wv = ee[i] * acc[t][i];
      unsafeAtomicAdd(&ent_num[(size_t)srcm[i] * OUTF + nn], wv);
      unsafeAtomicAdd(&typ_num[(size_t)dstm[i] * OUTF + nn], wv);
    }
  }
  if (lx == 0) {
    #pragma unroll
    for (int i = 0; i < 8; ++i) {
      unsafeAtomicAdd(&ent_den[srcm[i]], ee[i]);
      unsafeAtomicAdd(&typ_den[dstm[i]], ee[i]);
    }
  }
}

// ---------------------------------------------------------------------------
// Main: per wave32, a 32-edge (2 x 16) M-tile. B fragments double-buffered
// so ds_load of step s+1 issues before the WMMAs of step s (no WAR stall).
// ---------------------------------------------------------------------------
__launch_bounds__(256)
__global__ void edge_attn(const float* __restrict__ x1,
                          const float* __restrict__ x2,
                          const float* __restrict__ eemb,
                          const unsigned short* __restrict__ bws,
                          const float* __restrict__ a2,
                          const int* __restrict__ esrc,
                          const int* __restrict__ edst,
                          float* __restrict__ ent_num, float* __restrict__ ent_den,
                          float* __restrict__ typ_num, float* __restrict__ typ_den) {
  __shared__ __align__(16) unsigned short sB[80 * 32 * 16];  // 80 KB

  int tid = threadIdx.x;

  // Stage all B fragments into LDS (reused by all waves, all k-steps).
  {
    const f4* s = (const f4*)bws;
    f4*       d = (f4*)sB;
    #pragma unroll
    for (int i = 0; i < 20; ++i) d[tid + i * 256] = s[tid + i * 256];
  }
  __syncthreads();

  int lane = tid & 31;
  int w    = tid >> 5;
  int pair = blockIdx.x * 8 + w;
  if (pair >= NE / 32) return;
  int base = pair * 32;

  int half = lane >> 4;
  int lx   = lane & 15;
  const unsigned short* sBlane = sB + lane * 16;

  int src0 = esrc[base + lx],      dst0 = edst[base + lx];
  int src1 = esrc[base + 16 + lx], dst1 = edst[base + 16 + lx];

  float a2v[8];
  #pragma unroll
  for (int t = 0; t < 8; ++t) a2v[t] = a2[t * 16 + lx];

  v8f acc0[8] = {};
  v8f acc1[8] = {};

  FragU Bb[2];
  loadB(Bb[0], sBlane, 0);

  #pragma unroll
  for (int kk = 0; kk < 10; ++kk) {
    FragU A0 = buildA(x1, x2, eemb, src0, dst0, base + lx,      kk, half);
    FragU A1 = buildA(x1, x2, eemb, src1, dst1, base + 16 + lx, kk, half);
    #pragma unroll
    for (int t = 0; t < 8; ++t) {
      int step = kk * 8 + t;
      if (step < 79) loadB(Bb[(step + 1) & 1], sBlane, step + 1);
      const FragU& B = Bb[step & 1];
      acc0[t] = __builtin_amdgcn_wmma_f32_16x16x32_bf16(
          false, A0.bf, false, B.bf, (short)0, acc0[t], false, false);
      acc1[t] = __builtin_amdgcn_wmma_f32_16x16x32_bf16(
          false, A1.bf, false, B.bf, (short)0, acc1[t], false, false);
    }
  }

  epilogue(acc0, a2v, base,      half, lx, esrc, edst, ent_num, ent_den, typ_num, typ_den);
  epilogue(acc1, a2v, base + 16, half, lx, esrc, edst, ent_num, ent_den, typ_num, typ_den);
}

// ---------------------------------------------------------------------------
// Finalize: normalize + ELU, write [entity | type] concatenated f32 output.
// ---------------------------------------------------------------------------
__global__ void finalize(const float* __restrict__ ent_num,
                         const float* __restrict__ ent_den,
                         const float* __restrict__ typ_num,
                         const float* __restrict__ typ_den,
                         float* __restrict__ out) {
  long long idx = (long long)blockIdx.x * 256 + threadIdx.x;
  long long tot = (long long)(N1 + N2) * OUTF;
  if (idx >= tot) return;
  long long node = idx >> 7;
  int n = (int)(idx & 127);
  float num, den;
  if (node < N1) {
    num = ent_num[node * OUTF + n];
    den = ent_den[node];
  } else {
    long long tn = node - N1;
    num = typ_num[tn * OUTF + n];
    den = typ_den[tn];
  }
  if (den == 0.f) den = 1e-12f;
  float h = num / den;
  out[idx] = h > 0.f ? h : __expf(h) - 1.f;
}

extern "C" void kernel_launch(void* const* d_in, const int* in_sizes, int n_in,
                              void* d_out, int out_size, void* d_ws, size_t ws_size,
                              hipStream_t stream) {
  (void)in_sizes; (void)n_in; (void)out_size; (void)ws_size;
  const float* x1   = (const float*)d_in[0];
  const float* x2   = (const float*)d_in[1];
  const float* eemb = (const float*)d_in[2];
  const float* a    = (const float*)d_in[3];
  const float* a2   = (const float*)d_in[4];
  const int*   esrc = (const int*)d_in[5];
  const int*   edst = (const int*)d_in[6];

  char* ws = (char*)d_ws;
  unsigned short* bws = (unsigned short*)ws;                       // 81920 B
  float* ent_num = (float*)(ws + 81920);                           // N1*128 f32
  float* ent_den = (float*)((char*)ent_num + (size_t)N1 * OUTF * 4);
  float* typ_num = (float*)((char*)ent_den + (size_t)N1 * 4);
  float* typ_den = (float*)((char*)typ_num + (size_t)N2 * OUTF * 4);

  size_t accumBytes = (size_t)N1 * OUTF * 4 + (size_t)N1 * 4 +
                      (size_t)N2 * OUTF * 4 + (size_t)N2 * 4;
  hipMemsetAsync(ws + 81920, 0, accumBytes, stream);

  prep_bfrag<<<(80 * 32 * 16 + 255) / 256, 256, 0, stream>>>(a, bws);

  int pairs  = NE / 32;                 // 31250
  int blocks = (pairs + 7) / 8;         // 8 waves / block
  edge_attn<<<blocks, 256, 0, stream>>>(x1, x2, eemb, bws, a2, esrc, edst,
                                        ent_num, ent_den, typ_num, typ_den);

  long long tot = (long long)(N1 + N2) * OUTF;
  finalize<<<(int)((tot + 255) / 256), 256, 0, stream>>>(ent_num, ent_den,
                                                         typ_num, typ_den,
                                                         (float*)d_out);
}
